// NMMONet_40793599378040
// MI455X (gfx1250) — matrix operand order repaired and checked by reading, not compiled, not run on hardware
//
#include <hip/hip_runtime.h>
#include <hip/hip_bf16.h>
#include <math.h>

// ---------------------------------------------------------------------------
// Problem constants (from reference)
// ---------------------------------------------------------------------------
#define T_STEPS 16
#define BATCH   512
#define TB      (T_STEPS * BATCH)   // 8192
#define N_ENT   100
#define D_ENT   48
#define C_SCAT  48
#define HGT     15
#define WID     15
#define CELLS   (HGT * WID)         // 225
#define C_CONV  32
#define K_CONV  (3 * 3 * C_SCAT)    // 432
#define K_CONVP 448                 // padded to multiple of 32
#define D_IN    256
#define K_FC    (C_CONV * CELLS)    // 7200
#define H_LSTM  512
#define G4      (4 * H_LSTM)        // 2048

typedef __attribute__((ext_vector_type(16))) _Float16 v16h;
typedef __attribute__((ext_vector_type(8)))  _Float16 v8h;
typedef __attribute__((ext_vector_type(8)))  float    v8f;
typedef __attribute__((ext_vector_type(4)))  unsigned int u32x4;
typedef __attribute__((ext_vector_type(8)))  int          i32x8;
typedef __attribute__((ext_vector_type(4)))  int          i32x4;

#if __has_builtin(__builtin_amdgcn_tensor_load_to_lds) && \
    __has_builtin(__builtin_amdgcn_s_wait_tensorcnt)
#define USE_TDM 1
#else
#define USE_TDM 0
#endif

// ---------------------------------------------------------------------------
// WMMA fragment helpers (CDNA5 16x16x32 f16, wave32 layouts per ISA 7.12.2)
// ---------------------------------------------------------------------------
// A: 16x32 (MxK) row-major tile. Lane L<16 holds M=L, K in {0..7}U{16..23};
// lane L>=16 holds M=L-16, K in {8..15}U{24..31}.
__device__ __forceinline__ v16h load_a_frag(const _Float16* A, int lda) {
  const int lane = threadIdx.x & 31;
  const int m = lane & 15;
  const int kbase = (lane < 16) ? 0 : 8;
  const _Float16* row = A + m * lda;
  v16h a;
#pragma unroll
  for (int j = 0; j < 16; ++j) {
    const int kk = kbase + j + ((j >= 8) ? 8 : 0);
    a[j] = row[kk];
  }
  return a;
}

// B: 32x16 (KxN) row-major tile. Lane holds column N=lane&15;
// lanes 0..15 hold K in {0..7}U{16..23}, lanes 16..31 hold K in {8..15}U{24..31}.
__device__ __forceinline__ v16h load_b_frag(const _Float16* B, int ldb) {
  const int lane = threadIdx.x & 31;
  const int n = lane & 15;
  const int kbase = (lane < 16) ? 0 : 8;
  v16h b;
#pragma unroll
  for (int j = 0; j < 16; ++j) {
    b[j] = B[(kbase + j + ((j >= 8) ? 8 : 0)) * ldb + n];
  }
  return b;
}

__device__ __forceinline__ v8f wmma_f16(v16h a, v16h b, v8f c) {
  return __builtin_amdgcn_wmma_f32_16x16x32_f16(false, a, false, b, (short)0, c,
                                                false, false);
}

__device__ __forceinline__ float sigmoidf_(float x) {
  return 1.0f / (1.0f + __expf(-x));
}

#if USE_TDM
// ---------------------------------------------------------------------------
// Tensor Data Mover: async 2D tile load global->LDS (ISA ch.8 D# descriptor).
// rows x cols (f16 elements), row stride ld (elements). Wave-uniform args.
// ---------------------------------------------------------------------------
__device__ __forceinline__ void tdm_load_2d(const _Float16* gsrc, void* lds_dst,
                                            int rows, int cols, int ld,
                                            int tdim0, int tdim1) {
  const unsigned long long ga = (unsigned long long)(size_t)gsrc;
  u32x4 g0;
  g0[0] = 1u;                                     // count=1, user mode
  g0[1] = (unsigned)(size_t)lds_dst;              // lds_addr (byte)
  g0[2] = (unsigned)(ga & 0xffffffffu);           // global_addr[31:0]
  g0[3] = ((unsigned)(ga >> 32) & 0x01ffffffu) | (2u << 30);  // addr[56:32], type=2
  i32x8 g1;
  g1[0] = 1 << 16;                                // data_size=1 (2 bytes), no mask
  g1[1] = (int)(((unsigned)tdim0 & 0xffffu) << 16);         // tensor_dim0[15:0]
  g1[2] = (int)((((unsigned)tdim0 >> 16) & 0xffffu) |
                (((unsigned)tdim1 & 0xffffu) << 16));       // dim0 hi, dim1 lo
  g1[3] = (int)((((unsigned)tdim1 >> 16) & 0xffffu) |
                (((unsigned)cols & 0xffffu) << 16));        // dim1 hi, tile_dim0
  g1[4] = rows & 0xffff;                          // tile_dim1 (tile_dim2 = 0)
  g1[5] = ld;                                     // tensor_dim0_stride[31:0]
  g1[6] = 0;                                      // stride hi / dim1_stride lo
  g1[7] = 0;
  const i32x4 z4 = {0, 0, 0, 0};
#if __clang_major__ >= 23
  const i32x8 z8 = {0, 0, 0, 0, 0, 0, 0, 0};
  __builtin_amdgcn_tensor_load_to_lds(g0, g1, z4, z4, z8, 0);
#else
  __builtin_amdgcn_tensor_load_to_lds(g0, g1, z4, z4, 0);
#endif
}
#endif

// ---------------------------------------------------------------------------
// Utility kernels
// ---------------------------------------------------------------------------
__global__ void zero_f32(float* p, size_t n) {
  for (size_t i = (size_t)blockIdx.x * blockDim.x + threadIdx.x; i < n;
       i += (size_t)gridDim.x * blockDim.x)
    p[i] = 0.0f;
}

__global__ void zero_f16(_Float16* p, size_t n) {
  for (size_t i = (size_t)blockIdx.x * blockDim.x + threadIdx.x; i < n;
       i += (size_t)gridDim.x * blockDim.x)
    p[i] = (_Float16)0.0f;
}

__global__ void cvt_f32_f16(const float* in, _Float16* out, size_t n) {
  for (size_t i = (size_t)blockIdx.x * blockDim.x + threadIdx.x; i < n;
       i += (size_t)gridDim.x * blockDim.x)
    out[i] = (_Float16)in[i];
}

// conv_w HWIO [3,3,48,32] -> f16 [448][32], K zero-padded 432->448
__global__ void cvt_convw(const float* w, _Float16* out) {
  const int i = blockIdx.x * blockDim.x + threadIdx.x;
  if (i >= K_CONVP * C_CONV) return;
  const int k = i / C_CONV;
  out[i] = (k < K_CONV) ? (_Float16)w[i] : (_Float16)0.0f;
}

// W [N][K] row-major -> WT f16 [K][N] row-major (B operand of x @ W^T)
__global__ void cvt_transpose_f16(const float* in, _Float16* out, int K, int N) {
  const size_t total = (size_t)K * N;
  for (size_t i = (size_t)blockIdx.x * blockDim.x + threadIdx.x; i < total;
       i += (size_t)gridDim.x * blockDim.x) {
    const int k = (int)(i / N);
    const int n = (int)(i % N);
    out[i] = (_Float16)in[(size_t)n * K + k];
  }
}

// ---------------------------------------------------------------------------
// Stage 1+2: per-entity projection (relu(ef @ W + b)) + scatter-add
// ---------------------------------------------------------------------------
__global__ void proj_scatter(const float* __restrict__ ef,
                             const int* __restrict__ loc,
                             const float* __restrict__ W,
                             const float* __restrict__ bias,
                             float* __restrict__ smap) {
  __shared__ float sW[D_ENT * C_SCAT];
  __shared__ float sb[C_SCAT];
  const int tb = blockIdx.x;
  const int tid = threadIdx.x;
  for (int i = tid; i < D_ENT * C_SCAT; i += blockDim.x) sW[i] = W[i];
  if (tid < C_SCAT) sb[tid] = bias[tid];
  __syncthreads();

  const float* E = ef + (size_t)tb * N_ENT * D_ENT;
  const int* L = loc + (size_t)tb * N_ENT * 2;
  float* M = smap + (size_t)tb * CELLS * C_SCAT;

  for (int o = tid; o < N_ENT * C_SCAT; o += blockDim.x) {
    const int n = o / C_SCAT;
    const int c = o % C_SCAT;
    const float* e = E + n * D_ENT;
    float acc = sb[c];
#pragma unroll
    for (int k = 0; k < D_ENT; ++k) acc += e[k] * sW[k * C_SCAT + c];
    acc = fmaxf(acc, 0.0f);
    const int cell = L[n * 2] * WID + L[n * 2 + 1];
    atomicAdd(&M[(size_t)cell * C_SCAT + c], acc);
  }
}

// ---------------------------------------------------------------------------
// Stage 3: 3x3 conv as implicit GEMM, M=225(->240) N=32 K=432(->448), WMMA
// One block per TB sample. 8 waves, 30 tile-jobs (15 M-tiles x 2 N-tiles).
// ---------------------------------------------------------------------------
__global__ void conv_wmma(const float* __restrict__ smap,
                          const _Float16* __restrict__ w16,
                          const float* __restrict__ conv_b,
                          _Float16* __restrict__ y16) {
  __shared__ _Float16 s_in[17 * 17 * C_SCAT];  // padded input tile (halo=1)
  __shared__ _Float16 s_w[K_CONVP * C_CONV];   // all im2col weights
  const int tb = blockIdx.x;
  const int tid = threadIdx.x;
  const int lane = tid & 31;
  const int wave = tid >> 5;

  const float* M = smap + (size_t)tb * CELLS * C_SCAT;
  for (int i = tid; i < 17 * 17 * C_SCAT; i += blockDim.x) {
    const int y = i / (17 * C_SCAT);
    const int rem = i - y * 17 * C_SCAT;
    const int x = rem / C_SCAT;
    const int c = rem - x * C_SCAT;
    _Float16 v = (_Float16)0.0f;
    if (y >= 1 && y <= HGT && x >= 1 && x <= WID)
      v = (_Float16)M[((y - 1) * WID + (x - 1)) * C_SCAT + c];
    s_in[i] = v;
  }
  for (int i = tid; i < K_CONVP * C_CONV; i += blockDim.x) s_w[i] = w16[i];
  __syncthreads();

  _Float16* Y = y16 + (size_t)tb * K_FC;

  for (int job = wave; job < 30; job += 8) {
    const int mtile = job >> 1;
    const int ntile = job & 1;
    const int p = mtile * 16 + (lane & 15);
    const int kbase = (lane < 16) ? 0 : 8;
    const int py0 = (p < CELLS) ? (p / WID) : 0;
    const int px0 = (p < CELLS) ? (p % WID) : 0;

    v8f acc = {};
#pragma unroll 1
    for (int ks = 0; ks < K_CONVP / 32; ++ks) {
      const int k0 = ks * 32;
      v16h a;
#pragma unroll
      for (int j = 0; j < 16; ++j) {
        const int k = k0 + kbase + j + ((j >= 8) ? 8 : 0);
        _Float16 v = (_Float16)0.0f;
        if (p < CELLS && k < K_CONV) {
          const int ky = k / (3 * C_SCAT);
          const int r = k - ky * 3 * C_SCAT;
          const int kx = r / C_SCAT;
          const int ci = r - kx * C_SCAT;
          v = s_in[((py0 + ky) * 17 + (px0 + kx)) * C_SCAT + ci];
        }
        a[j] = v;
      }
      const v16h b = load_b_frag(s_w + k0 * C_CONV + ntile * 16, C_CONV);
      acc = wmma_f16(a, b, acc);
    }
    const int co = ntile * 16 + (lane & 15);
    const float bv = conv_b[co];
    const int prow0 = mtile * 16 + ((lane >= 16) ? 8 : 0);
#pragma unroll
    for (int r = 0; r < 8; ++r) {
      const int pr = prow0 + r;
      if (pr < CELLS)
        Y[(size_t)pr * C_CONV + co] = (_Float16)fmaxf(acc[r] + bv, 0.0f);
    }
  }
}

// ---------------------------------------------------------------------------
// Generic WMMA GEMM: C[MxN] = A[MxK] * B[KxN]  (f16 in, f32 accumulate)
// Block tile 64x128, 8 waves; each wave a 16x64 strip (4 accumulators,
// A-fragment reused 4x). Double-buffered LDS fed by TDM async tile loads
// when available (TENSORcnt-fenced), else synchronous vector loads.
// Requires M%64==0, N%128==0, K%32==0.
// EPI==0: fp32 store.  EPI==1: bias+relu, f16 store.
// ---------------------------------------------------------------------------
template <int EPI>
__global__ void gemm_wmma(const _Float16* __restrict__ A,
                          const _Float16* __restrict__ B, void* __restrict__ Cout,
                          const float* __restrict__ bias, int M, int N, int K) {
  __shared__ _Float16 sA[2][64 * 32];
  __shared__ _Float16 sB[2][32 * 128];
  const int tid = threadIdx.x;
  const int lane = tid & 31;
  const int wave = tid >> 5;
  const int m0 = blockIdx.x * 64;
  const int n0 = blockIdx.y * 128;
  const int wm = (wave >> 1) * 16;  // 0,16,32,48
  const int wn = (wave & 1) * 64;   // 0,64
  v8f acc[4] = {};
  const int nsteps = K / 32;

#if USE_TDM
  if (wave == 0) {
    tdm_load_2d(A + (size_t)m0 * K, sA[0], 64, 32, K, K, M);
    tdm_load_2d(B + n0, sB[0], 32, 128, N, N, K);
  }
#pragma unroll 1
  for (int s = 0; s < nsteps; ++s) {
    const int buf = s & 1;
    if (wave == 0) __builtin_amdgcn_s_wait_tensorcnt(0);
    __syncthreads();  // tiles[s] visible; compute of tiles[s-1] finished
    if (wave == 0 && (s + 1) < nsteps) {
      const int k1 = (s + 1) * 32;
      tdm_load_2d(A + (size_t)m0 * K + k1, sA[buf ^ 1], 64, 32, K, K, M);
      tdm_load_2d(B + (size_t)k1 * N + n0, sB[buf ^ 1], 32, 128, N, N, K);
    }
    const v16h a = load_a_frag(sA[buf] + wm * 32, 32);
#pragma unroll
    for (int t = 0; t < 4; ++t) {
      const v16h b = load_b_frag(sB[buf] + wn + t * 16, 128);
      acc[t] = wmma_f16(a, b, acc[t]);
    }
  }
#else
  const int ar = tid >> 2, ac = (tid & 3) * 8;    // A: 64 rows x 4 x v8h
  const int br = tid >> 3, bc = (tid & 7) * 16;   // B: 32 rows x 8 x 2*v8h
#pragma unroll 1
  for (int s = 0; s < nsteps; ++s) {
    const int k0 = s * 32;
    *(v8h*)(sA[0] + ar * 32 + ac) =
        *(const v8h*)(A + (size_t)(m0 + ar) * K + k0 + ac);
    *(v8h*)(sB[0] + br * 128 + bc) =
        *(const v8h*)(B + (size_t)(k0 + br) * N + n0 + bc);
    *(v8h*)(sB[0] + br * 128 + bc + 8) =
        *(const v8h*)(B + (size_t)(k0 + br) * N + n0 + bc + 8);
    if (k0 + 32 < K) {
      __builtin_prefetch(A + (size_t)(m0 + ar) * K + k0 + 32 + ac, 0, 1);
      __builtin_prefetch(B + (size_t)(k0 + 32 + br) * N + n0 + bc, 0, 1);
    }
    __syncthreads();
    const v16h a = load_a_frag(sA[0] + wm * 32, 32);
#pragma unroll
    for (int t = 0; t < 4; ++t) {
      const v16h b = load_b_frag(sB[0] + wn + t * 16, 128);
      acc[t] = wmma_f16(a, b, acc[t]);
    }
    __syncthreads();
  }
#endif

  const int row0 = m0 + wm + ((lane >= 16) ? 8 : 0);
  const int col = lane & 15;
#pragma unroll
  for (int t = 0; t < 4; ++t) {
    const int cc = n0 + wn + t * 16 + col;
    if (EPI == 0) {
      float* C = (float*)Cout;
#pragma unroll
      for (int r = 0; r < 8; ++r)
        C[(size_t)(row0 + r) * N + cc] = acc[t][r];
    } else {
      _Float16* C = (_Float16*)Cout;
      const float bv = bias ? bias[cc] : 0.0f;
#pragma unroll
      for (int r = 0; r < 8; ++r)
        C[(size_t)(row0 + r) * N + cc] = (_Float16)fmaxf(acc[t][r] + bv, 0.0f);
    }
  }
}

// ---------------------------------------------------------------------------
// Stage 5 pointwise: dual LayerNorm over 2048 gates, LSTM cell update,
// LayerNorm over 512 cell state. One block (256 thr) per batch row.
// ---------------------------------------------------------------------------
__device__ __forceinline__ float block_sum(float v, float* red) {
  const int tid = threadIdx.x;
  red[tid] = v;
  __syncthreads();
  for (int s = 128; s > 0; s >>= 1) {
    if (tid < s) red[tid] += red[tid + s];
    __syncthreads();
  }
  const float r = red[0];
  __syncthreads();
  return r;
}

__global__ void lstm_cell(const float* __restrict__ G1, const float* __restrict__ G2,
                          const float* __restrict__ gi, const float* __restrict__ bi,
                          const float* __restrict__ gh, const float* __restrict__ bh,
                          const float* __restrict__ gc, const float* __restrict__ bc,
                          float* __restrict__ c_state, _Float16* __restrict__ h16,
                          float* __restrict__ hout) {
  __shared__ float red[256];
  __shared__ float s_c[H_LSTM];
  __shared__ float s_o[H_LSTM];
  const int b = blockIdx.x;
  const int tid = threadIdx.x;
  const float* r1 = G1 + (size_t)b * G4;
  const float* r2 = G2 + (size_t)b * G4;

  float s = 0.f, ss = 0.f;
  for (int j = tid; j < G4; j += 256) { const float v = r1[j]; s += v; ss += v * v; }
  const float m1 = block_sum(s, red) * (1.0f / G4);
  const float rs1 = rsqrtf(block_sum(ss, red) * (1.0f / G4) - m1 * m1 + 1e-5f);

  s = 0.f; ss = 0.f;
  for (int j = tid; j < G4; j += 256) { const float v = r2[j]; s += v; ss += v * v; }
  const float m2 = block_sum(s, red) * (1.0f / G4);
  const float rs2 = rsqrtf(block_sum(ss, red) * (1.0f / G4) - m2 * m2 + 1e-5f);

  for (int k = tid; k < H_LSTM; k += 256) {
    const int ji = k, jf = k + H_LSTM, jg = k + 2 * H_LSTM, jo = k + 3 * H_LSTM;
    const float vi = (r1[ji] - m1) * rs1 * gi[ji] + bi[ji] + (r2[ji] - m2) * rs2 * gh[ji] + bh[ji];
    const float vf = (r1[jf] - m1) * rs1 * gi[jf] + bi[jf] + (r2[jf] - m2) * rs2 * gh[jf] + bh[jf];
    const float vg = (r1[jg] - m1) * rs1 * gi[jg] + bi[jg] + (r2[jg] - m2) * rs2 * gh[jg] + bh[jg];
    const float vo = (r1[jo] - m1) * rs1 * gi[jo] + bi[jo] + (r2[jo] - m2) * rs2 * gh[jo] + bh[jo];
    s_c[k] = sigmoidf_(vf) * c_state[(size_t)b * H_LSTM + k] + sigmoidf_(vi) * tanhf(vg);
    s_o[k] = vo;
  }
  __syncthreads();

  s = 0.f; ss = 0.f;
  for (int k = tid; k < H_LSTM; k += 256) { const float v = s_c[k]; s += v; ss += v * v; }
  const float mc = block_sum(s, red) * (1.0f / H_LSTM);
  const float rsc = rsqrtf(block_sum(ss, red) * (1.0f / H_LSTM) - mc * mc + 1e-5f);

  for (int k = tid; k < H_LSTM; k += 256) {
    const float cn = (s_c[k] - mc) * rsc * gc[k] + bc[k];
    c_state[(size_t)b * H_LSTM + k] = cn;
    const float h = sigmoidf_(s_o[k]) * tanhf(cn);
    hout[(size_t)b * H_LSTM + k] = h;
    h16[(size_t)b * H_LSTM + k] = (_Float16)h;
  }
}

// ---------------------------------------------------------------------------
// Launch
// ---------------------------------------------------------------------------
extern "C" void kernel_launch(void* const* d_in, const int* in_sizes, int n_in,
                              void* d_out, int out_size, void* d_ws, size_t ws_size,
                              hipStream_t stream) {
  (void)in_sizes; (void)n_in; (void)out_size; (void)ws_size;
  const float* entity_feats = (const float*)d_in[0];
  const int*   entity_loc   = (const int*)d_in[1];
  const float* W_proj       = (const float*)d_in[2];
  const float* b_proj       = (const float*)d_in[3];
  const float* conv_w       = (const float*)d_in[4];
  const float* conv_b       = (const float*)d_in[5];
  const float* fc_w         = (const float*)d_in[6];
  const float* fc_b         = (const float*)d_in[7];
  const float* w_ih         = (const float*)d_in[8];
  const float* w_hh         = (const float*)d_in[9];
  const float* g_i          = (const float*)d_in[10];
  const float* bta_i        = (const float*)d_in[11];
  const float* g_h          = (const float*)d_in[12];
  const float* bta_h        = (const float*)d_in[13];
  const float* g_c          = (const float*)d_in[14];
  const float* bta_c        = (const float*)d_in[15];
  float* out = (float*)d_out;

  char* p = (char*)d_ws;
  auto alloc = [&](size_t bytes) -> void* {
    void* r = (void*)p;
    p += (bytes + 255) & ~(size_t)255;
    return r;
  };
  float*     smap   = (float*)alloc((size_t)TB * CELLS * C_SCAT * 4);
  _Float16*  y16    = (_Float16*)alloc((size_t)TB * K_FC * 2);
  _Float16*  x16    = (_Float16*)alloc((size_t)TB * D_IN * 2);
  _Float16*  cw16   = (_Float16*)alloc((size_t)K_CONVP * C_CONV * 2);
  _Float16*  fcw16  = (_Float16*)alloc((size_t)K_FC * D_IN * 2);
  _Float16*  wihT16 = (_Float16*)alloc((size_t)D_IN * G4 * 2);
  _Float16*  whhT16 = (_Float16*)alloc((size_t)H_LSTM * G4 * 2);
  float*     G1     = (float*)alloc((size_t)BATCH * G4 * 4);
  float*     G2     = (float*)alloc((size_t)BATCH * G4 * 4);
  _Float16*  h16    = (_Float16*)alloc((size_t)BATCH * H_LSTM * 2);
  float*     cst    = (float*)alloc((size_t)BATCH * H_LSTM * 4);

  zero_f32<<<4096, 256, 0, stream>>>(smap, (size_t)TB * CELLS * C_SCAT);
  zero_f16<<<1024, 256, 0, stream>>>(h16, (size_t)BATCH * H_LSTM);
  zero_f32<<<1024, 256, 0, stream>>>(cst, (size_t)BATCH * H_LSTM);
  {
    const int n = K_CONVP * C_CONV;
    cvt_convw<<<(n + 255) / 256, 256, 0, stream>>>(conv_w, cw16);
  }
  cvt_f32_f16<<<4096, 256, 0, stream>>>(fc_w, fcw16, (size_t)K_FC * D_IN);
  cvt_transpose_f16<<<2048, 256, 0, stream>>>(w_ih, wihT16, D_IN, G4);
  cvt_transpose_f16<<<4096, 256, 0, stream>>>(w_hh, whhT16, H_LSTM, G4);

  proj_scatter<<<TB, 256, 0, stream>>>(entity_feats, entity_loc, W_proj, b_proj, smap);
  conv_wmma<<<TB, 256, 0, stream>>>(smap, cw16, conv_b, y16);

  // FC 7200->256 (M=8192 N=256 K=7200), bias+relu, f16 out
  gemm_wmma<1><<<dim3(TB / 64, D_IN / 128), 256, 0, stream>>>(
      y16, fcw16, (void*)x16, fc_b, TB, D_IN, K_FC);

  for (int t = 0; t < T_STEPS; ++t) {
    const _Float16* xt = x16 + (size_t)t * BATCH * D_IN;
    gemm_wmma<0><<<dim3(BATCH / 64, G4 / 128), 256, 0, stream>>>(
        xt, wihT16, (void*)G1, nullptr, BATCH, G4, D_IN);
    gemm_wmma<0><<<dim3(BATCH / 64, G4 / 128), 256, 0, stream>>>(
        h16, whhT16, (void*)G2, nullptr, BATCH, G4, H_LSTM);
    lstm_cell<<<BATCH, 256, 0, stream>>>(G1, G2, g_i, bta_i, g_h, bta_h, g_c, bta_c,
                                         cst, h16, out + (size_t)t * BATCH * H_LSTM);
  }
}
